// mslstm_48713519071849
// MI455X (gfx1250) — matrix operand order repaired and checked by reading, not compiled, not run on hardware
//
#include <hip/hip_runtime.h>
#include <stdint.h>

// ---------------------------------------------------------------------------
// MI455X (gfx1250) fused msLSTM step.
// Big GEMM: M=16384 (B*S), K=2048 ([ch, bef_ch, aft_ch, vec]),
//           N=3584 (7 gate columns interleaved per feature: c = h*7 + g).
// bf16 WMMA 16x16x32, f32 accumulation. Operands pre-packed into exact WMMA
// fragment layouts: hot loop is pure global_load_b128 + v_wmma.
// Each wave owns 2 M-fragments x 7 N-fragments -> B fragments amortized 2x.
// Note: WGP-scope asm prefetch was tried and REVERTED — asm volatile acts as a
// scheduler barrier and cost the x2 unroll / load-WMMA overlap; operands are
// GL2-resident anyway, so schedulable builtin prefetch wins.
// ---------------------------------------------------------------------------

typedef __attribute__((ext_vector_type(16))) __bf16 v16bf;
typedef __attribute__((ext_vector_type(8)))  float  v8f;

#define B_    8
#define S_    2048
#define H_    512
#define K_    2048            // 4*H
#define NCOL  3584            // 7*H, interleaved c = h*7 + g
#define BS_   (B_ * S_)       // 16384
#define MTB   128             // rows per workgroup (8 m-fragments)
#define NT    224             // cols per workgroup (32 features x 7 gates)
#define KTW   32              // K per WMMA
#define KSTEPS (K_ / KTW)     // 64
#define MTILES (BS_ / MTB)    // 128
#define NTILES (NCOL / NT)    // 16

__device__ __forceinline__ unsigned short f2bf(float f) {
  union { float f; uint32_t u; } v; v.f = f;
  uint32_t u = v.u;
  u += 0x7fffu + ((u >> 16) & 1u);   // round-to-nearest-even
  return (unsigned short)(u >> 16);
}

__device__ __forceinline__ float sigf(float x) {
  return 1.0f / (1.0f + __expf(-x));
}

// --------------------------- pack X (A fragments) ---------------------------
// Xpack[mf(1024)][kt(64)][lane(32)][16 bf16], A 16-bit 16x32 fragment layout:
// lane: m = 16*mf + lane%16, half = lane/16;
// elem e: j=e/2, lo=e&1 -> k = kt*32 + (j>=4?16:0) + 8*half + 2*(j&3) + lo
struct XPtrs { const float* ch; const float* vec; };

__global__ void pack_x_kernel(XPtrs p, unsigned short* __restrict__ xp) {
  const int tid  = blockIdx.x * 256 + threadIdx.x;   // 2,097,152 threads
  const int lane = tid & 31;
  const int fg   = tid >> 5;                         // 0..65535
  const int kt   = fg & 63;
  const int mf   = fg >> 6;                          // 0..1023
  const int m    = mf * 16 + (lane & 15);
  const int hf   = lane >> 4;
  const int b    = m >> 11;
  const int s    = m & (S_ - 1);
  const float* chb = p.ch + (size_t)b * S_ * H_;
  unsigned short* out = xp + ((size_t)fg * 32 + lane) * 16;
#pragma unroll
  for (int e = 0; e < 16; ++e) {
    const int j = e >> 1, lo = e & 1;
    const int k = kt * 32 + ((j & 4) ? 16 : 0) + 8 * hf + 2 * (j & 3) + lo;
    float val;
    if (k < 512) {
      val = chb[(size_t)s * H_ + k];
    } else if (k < 1024) {                   // bef_ch = ch[s-1] + ch[s-2]
      const int kk = k - 512;
      val = (s >= 1 ? chb[(size_t)(s - 1) * H_ + kk] : 0.f)
          + (s >= 2 ? chb[(size_t)(s - 2) * H_ + kk] : 0.f);
    } else if (k < 1536) {                   // aft_ch = ch[s+1] + ch[s+2]
      const int kk = k - 1024;
      val = (s + 1 < S_ ? chb[(size_t)(s + 1) * H_ + kk] : 0.f)
          + (s + 2 < S_ ? chb[(size_t)(s + 2) * H_ + kk] : 0.f);
    } else {
      val = p.vec[((size_t)b * S_ + s) * H_ + (k - 1536)];
    }
    out[e] = f2bf(val);
  }
}

// --------------------------- pack W (B fragments) ---------------------------
// Wpack[nf(224)][kt(64)][lane(32)][16 bf16], B 16-bit 32x16 fragment layout:
// lane: n = 16*nf + lane%16, half = lane/16; elem e: k = kt*32 + (e>=8?16:0) + 8*half + (e&7).
// Column c = n -> feature h = c/7, gate g = c%7 (g=0..5 -> f1..f4,fi,fo; g=6 gft)
struct WPtrs {
  const float* blk0[7];   // K 0..511    : tx_w (g<6) / gfth_w (g==6)
  const float* th[6];     // K 512..1535 : th_w rows 0..1023
  const float* ti[6];     // K 1536..2047: ti_w
};

__global__ void pack_w_kernel(WPtrs p, unsigned short* __restrict__ wp) {
  const int tid  = blockIdx.x * 256 + threadIdx.x;   // 458,752 threads
  const int lane = tid & 31;
  const int fg   = tid >> 5;                         // 0..14335
  const int kt   = fg & 63;
  const int nf   = fg >> 6;                          // 0..223
  const int n    = nf * 16 + (lane & 15);
  const int h    = n / 7, g = n % 7;
  const int hf   = lane >> 4;
  unsigned short* out = wp + ((size_t)fg * 32 + lane) * 16;
#pragma unroll
  for (int e = 0; e < 16; ++e) {
    const int k = kt * 32 + ((e & 8) ? 16 : 0) + 8 * hf + (e & 7);
    float val = 0.f;
    if (k < 512)            val = p.blk0[g][(size_t)k * H_ + h];
    else if (k < 1536) { if (g < 6) val = p.th[g][(size_t)(k - 512)  * H_ + h]; }
    else               { if (g < 6) val = p.ti[g][(size_t)(k - 1536) * H_ + h]; }
    out[e] = f2bf(val);
  }
}

// --------------------------- c_ch = mean_s(ch) ------------------------------
__global__ void cch_kernel(const float* __restrict__ ch, float* __restrict__ c_ch) {
  const int t = blockIdx.x * 256 + threadIdx.x;      // 4096
  const int b = t >> 9, h = t & (H_ - 1);
  const float* base = ch + (size_t)b * S_ * H_ + h;
  float acc = 0.f;
  for (int s = 0; s < S_; ++s) acc += base[(size_t)s * H_];
  c_ch[t] = acc * (1.0f / S_);
}

// ---- per-batch column bias: b_tx+b_th+b_ti+b_td + dh@W_td  (g==6: gft) -----
struct BiasPtrs {
  const float* tx_b[6]; const float* th_b[6]; const float* ti_b[6];
  const float* td_w[6]; const float* td_b[6];
  const float* gftx_w; const float* gftx_b; const float* gfth_b;
  const float* dh;
};

__global__ void bias_kernel(BiasPtrs p, float* __restrict__ bias) {
  const int t = blockIdx.x * 256 + threadIdx.x;      // 28672
  const int b = t / NCOL, c = t % NCOL;
  const int h = c / 7, g = c % 7;
  const float* dh = p.dh + b * H_;
  float acc;
  const float* w;
  if (g < 6) {
    acc = p.tx_b[g][h] + p.th_b[g][h] + p.ti_b[g][h] + p.td_b[g][h];
    w = p.td_w[g];
  } else {
    acc = p.gftx_b[h] + p.gfth_b[h];
    w = p.gftx_w;
  }
  for (int k = 0; k < H_; ++k) acc += dh[k] * w[(size_t)k * H_ + h];
  bias[t] = acc;
}

// ------------- gdt pre-activation + n_dh (tiny per-batch matmuls) -----------
struct SmallPtrs {
  const float* gdtx_w; const float* gdtx_b;
  const float* gotx_w; const float* gotx_b;
  const float* goth_w; const float* goth_b;
  const float* dh; const float* dc; const float* c_ch;
};

__global__ void small_kernel(SmallPtrs p, float* __restrict__ gdtpre,
                             float* __restrict__ ndh_out) {
  const int t = blockIdx.x * 256 + threadIdx.x;      // 4096
  const int b = t >> 9, h = t & (H_ - 1);
  const float* dh = p.dh + b * H_;
  const float* cm = p.c_ch + b * H_;
  float gd = 2.0f * p.gdtx_b[h];                     // reference uses gdtx twice
  float go = p.gotx_b[h] + p.goth_b[h];
  for (int k = 0; k < H_; ++k) {
    const size_t wi = (size_t)k * H_ + h;
    gd += (dh[k] + cm[k]) * p.gdtx_w[wi];
    go += dh[k] * p.gotx_w[wi] + cm[k] * p.goth_w[wi];
  }
  gdtpre[t] = gd;
  ndh_out[t] = sigf(go) * tanhf(p.dc[t]);
}

// ------------------------------- main GEMM ----------------------------------
// Workgroup tile: 128 rows x 224 cols. 8 waves; wave w = (mp, ng):
//   mp = w>>1 owns m-fragments mp and mp+4 (one in each 64-row half),
//   ng = w&1 owns n-fragments ng*7 .. ng*7+6.
// Epilogue runs twice (64-row halves) so the LDS preact tile stays at 57 KB.
__global__ void __launch_bounds__(256)
gemm_kernel(const __bf16* __restrict__ xp, const __bf16* __restrict__ wp,
            const float* __restrict__ bias,
            const float* __restrict__ cc, const float* __restrict__ vec,
            const float* __restrict__ dc,
            float* __restrict__ out_cc, float* __restrict__ out_ch,
            float* __restrict__ gsum, float* __restrict__ gcc) {
  __shared__ float spre[64 * NT];        // 57,344 B preactivation half-tile
  __shared__ float red[2][8][32];        //  2,048 B gft partial reduction

  const int ntile = blockIdx.x;          // 0..15
  const int mtile = blockIdx.y;          // 0..127
  const int tid   = threadIdx.x;
  const int lane  = tid & 31;
  const int w     = tid >> 5;            // wave 0..7
  const int mp    = w >> 1;              // 0..3
  const int ng    = w & 1;               // 0..1

  const v8f zero = {0.f, 0.f, 0.f, 0.f, 0.f, 0.f, 0.f, 0.f};
  v8f acc[2][7];
#pragma unroll
  for (int hh = 0; hh < 2; ++hh)
#pragma unroll
    for (int j = 0; j < 7; ++j) acc[hh][j] = zero;

  // global 16-row fragment indices for this wave's two strips
  const size_t mfgA = (size_t)mtile * 8 + mp;        // rows half 0
  const size_t mfgB = (size_t)mtile * 8 + mp + 4;    // rows half 1
  const size_t nf0  = (size_t)ntile * 14 + ng * 7;

  for (int kt = 0; kt < KSTEPS; ++kt) {
    const v16bf a0 =
        *reinterpret_cast<const v16bf*>(xp + (((mfgA * 64 + kt) * 32) + lane) * 16);
    const v16bf a1 =
        *reinterpret_cast<const v16bf*>(xp + (((mfgB * 64 + kt) * 32) + lane) * 16);
    if (kt + 1 < KSTEPS) {  // schedulable prefetch of next A fragments
      __builtin_prefetch(xp + (((mfgA * 64 + kt + 1) * 32) + lane) * 16, 0, 3);
      __builtin_prefetch(xp + (((mfgB * 64 + kt + 1) * 32) + lane) * 16, 0, 3);
    }
#pragma unroll
    for (int j = 0; j < 7; ++j) {
      const v16bf bf = *reinterpret_cast<const v16bf*>(
          wp + ((((nf0 + j) * 64 + kt) * 32) + lane) * 16);
      acc[0][j] = __builtin_amdgcn_wmma_f32_16x16x32_bf16(
          false, a0, false, bf, (short)0, acc[0][j], false, false);
      acc[1][j] = __builtin_amdgcn_wmma_f32_16x16x32_bf16(
          false, a1, false, bf, (short)0, acc[1][j], false, false);
    }
  }

  // ---- two 64-row halves: spill C to LDS, then fused epilogue --------------
  const int hf  = lane >> 4, nlo = lane & 15;
  const int hl  = tid & 31;              // local feature for epilogue
  const int rg  = tid >> 5;              // row group (8 rows) for epilogue
  const int b   = (mtile * MTB) >> 11;   // batch (tile never crosses batches)
  const int h   = ntile * 32 + hl;       // global feature
  const float* ccb  = cc + (size_t)b * S_ * H_ + h;
  const float* brow = bias + (size_t)b * NCOL + (size_t)h * 7;
  const float dcv = dc[b * H_ + h];

  for (int hh = 0; hh < 2; ++hh) {
    if (hh) __syncthreads();             // previous half fully consumed
    // C fragment (16x16 f32): VGPR i holds M=i (lanes 0-15) / M=8+i (16-31),
    // N = lane%16.
#pragma unroll
    for (int j = 0; j < 7; ++j) {
      const int col = (ng * 7 + j) * 16 + nlo;
#pragma unroll
      for (int i = 0; i < 8; ++i) {
        const int row = mp * 16 + i + 8 * hf;
        spre[row * NT + col] = acc[hh][j][i];
      }
    }
    __syncthreads();

    const int row0 = mtile * MTB + hh * 64;
    const int s0   = row0 & (S_ - 1);
    float psum = 0.f, pcc = 0.f;
    for (int q = 0; q < 8; ++q) {
      const int r = rg * 8 + q;
      const int s = s0 + r;
      const float* pr = &spre[r * NT + hl * 7];
      // 5-way softmax over sigmoid(f1..f4, fi)
      float e0 = __expf(sigf(pr[0] + brow[0]));
      float e1 = __expf(sigf(pr[1] + brow[1]));
      float e2 = __expf(sigf(pr[2] + brow[2]));
      float e3 = __expf(sigf(pr[3] + brow[3]));
      float e4 = __expf(sigf(pr[4] + brow[4]));
      const float inv = 1.0f / (e0 + e1 + e2 + e3 + e4);
      const float ccv = ccb[(size_t)s * H_];
      const float bef = (s >= 1 ? ccb[(size_t)(s - 1) * H_] : 0.f)
                      + (s >= 2 ? ccb[(size_t)(s - 2) * H_] : 0.f);
      const float aft = (s + 1 < S_ ? ccb[(size_t)(s + 1) * H_] : 0.f)
                      + (s + 2 < S_ ? ccb[(size_t)(s + 2) * H_] : 0.f);
      const float vv  = vec[((size_t)b * S_ + s) * H_ + h];
      const float ccn = inv * (e0 * bef + e1 * aft + e2 * vv + e3 * dcv + e4 * ccv);
      const float chn = sigf(pr[5] + brow[5]) * tanhf(ccn);
      const size_t oi = ((size_t)b * S_ + s) * H_ + h;
      out_cc[oi] = ccn;
      out_ch[oi] = chn;
      // gft contribution for the S-wide softmax (deterministic partials)
      const float ge = __expf(sigf(pr[6] + brow[6]));
      psum += ge;
      pcc  += ge * ccv;
    }
    red[0][rg][hl] = psum;
    red[1][rg][hl] = pcc;
    __syncthreads();
    if (tid < 32) {
      float a0s = 0.f, a1s = 0.f;
#pragma unroll
      for (int g8 = 0; g8 < 8; ++g8) { a0s += red[0][g8][tid]; a1s += red[1][g8][tid]; }
      const size_t chunk = (size_t)mtile * 2 + hh;   // 64-row chunk id, 0..255
      const size_t gi = chunk * H_ + ntile * 32 + tid;
      gsum[gi] = a0s;
      gcc[gi]  = a1s;
    }
  }
}

// ------------------------------ n_dc finalize -------------------------------
__global__ void final_kernel(const float* __restrict__ gsum,
                             const float* __restrict__ gcc,
                             const float* __restrict__ gdtpre,
                             const float* __restrict__ dc,
                             float* __restrict__ ndc_out) {
  const int t = blockIdx.x * 256 + threadIdx.x;      // 4096
  const int b = t >> 9, h = t & (H_ - 1);
  float Z = 0.f, A = 0.f;
  for (int mt = 0; mt < 32; ++mt) {
    const size_t i = (size_t)(b * 32 + mt) * H_ + h;
    Z += gsum[i];
    A += gcc[i];
  }
  const float ged = __expf(sigf(gdtpre[t]));
  Z += ged;
  const float invZ = 1.0f / Z;
  ndc_out[t] = A * invZ * (1.0f / S_) + ged * invZ * dc[t];
}

// ------------------------------- launcher -----------------------------------
extern "C" void kernel_launch(void* const* d_in, const int* in_sizes, int n_in,
                              void* d_out, int out_size, void* d_ws, size_t ws_size,
                              hipStream_t stream) {
  (void)in_sizes; (void)n_in; (void)out_size; (void)ws_size;

  // Input order (setup_inputs dict order, params nested in insertion order,
  // each mk() contributing w then b):
  // 0:vec 1:cc 2:ch 3:dc 4:dh
  // 5,6:gdtx  7,8:gotx  9,10:goth  11,12:gftx  13,14:gfth
  // gate g in {f1,f2,f3,f4,fi,fo}: base 15+8g -> tx(w,b), th(w,b), ti(w,b), td(w,b)
  const float* vec = (const float*)d_in[0];
  const float* cc  = (const float*)d_in[1];
  const float* ch  = (const float*)d_in[2];
  const float* dc  = (const float*)d_in[3];
  const float* dh  = (const float*)d_in[4];
  auto P = [&](int i) { return (const float*)d_in[i]; };

  // Workspace layout (≈79.2 MB total)
  char* ws = (char*)d_ws;
  unsigned short* xpack = (unsigned short*)(ws);                       // 67,108,864 B
  unsigned short* wpack = (unsigned short*)(ws + (size_t)67108864);    // 14,680,064 B
  float* bias   = (float*)(ws + (size_t)81788928);                     //    114,688 B
  float* c_ch   = (float*)(ws + (size_t)81903616);                     //     16,384 B
  float* gdtpre = (float*)(ws + (size_t)81920000);                     //     16,384 B
  float* gsum   = (float*)(ws + (size_t)81936384);                     //    524,288 B
  float* gcc    = (float*)(ws + (size_t)82460672);                     //    524,288 B

  float* out      = (float*)d_out;
  float* out_cc   = out;
  float* out_ch   = out + (size_t)BS_ * H_;
  float* out_ndc  = out + 2 * (size_t)BS_ * H_;
  float* out_ndh  = out_ndc + B_ * H_;

  XPtrs xpt{ch, vec};
  pack_x_kernel<<<8192, 256, 0, stream>>>(xpt, xpack);

  WPtrs wpt;
  for (int g = 0; g < 6; ++g) {
    wpt.blk0[g] = P(15 + 8 * g + 0);   // tx_w
    wpt.th[g]   = P(15 + 8 * g + 2);   // th_w (1024x512)
    wpt.ti[g]   = P(15 + 8 * g + 4);   // ti_w
  }
  wpt.blk0[6] = P(13);                 // gfth_w
  pack_w_kernel<<<1792, 256, 0, stream>>>(wpt, wpack);

  cch_kernel<<<16, 256, 0, stream>>>(ch, c_ch);

  BiasPtrs bpt;
  for (int g = 0; g < 6; ++g) {
    bpt.tx_b[g] = P(15 + 8 * g + 1);
    bpt.th_b[g] = P(15 + 8 * g + 3);
    bpt.ti_b[g] = P(15 + 8 * g + 5);
    bpt.td_w[g] = P(15 + 8 * g + 6);
    bpt.td_b[g] = P(15 + 8 * g + 7);
  }
  bpt.gftx_w = P(11); bpt.gftx_b = P(12); bpt.gfth_b = P(14); bpt.dh = dh;
  bias_kernel<<<112, 256, 0, stream>>>(bpt, bias);

  SmallPtrs spt{P(5), P(6), P(7), P(8), P(9), P(10), dh, dc, c_ch};
  small_kernel<<<16, 256, 0, stream>>>(spt, gdtpre, out_ndh);

  dim3 grid(NTILES, MTILES);  // (16, 128)
  gemm_kernel<<<grid, 256, 0, stream>>>((const __bf16*)xpack, (const __bf16*)wpack,
                                        bias, cc, vec, dc,
                                        out_cc, out_ch, gsum, gcc);

  final_kernel<<<16, 256, 0, stream>>>(gsum, gcc, gdtpre, dc, out_ndc);
}
